// SpikingNeuralNetwork_54855322304664
// MI455X (gfx1250) — compile-verified, matrix operand order checked by reading, not run on hardware
//
#include <hip/hip_runtime.h>

// CDNA5 / gfx1250 SNN implementation.
// fp16 WMMA (V_WMMA_F32_16X16X32_F16) GEMMs with fused LIF epilogue.
// Weights pre-packed into wave32 WMMA B-operand tile layout (1KB coalesced
// blocks); spikes kept as fp16 0/1 (exact). K/N are compile-time template
// params so the 64-chunk K-loop has no remainder path and unrolls deeply.

typedef __attribute__((ext_vector_type(16))) _Float16 v16h;
typedef __attribute__((ext_vector_type(8)))  _Float16 v8h;
typedef __attribute__((ext_vector_type(8)))  float    v8f;

#define BATCH  32
#define TSTEPS 100
#define NIN    2048
#define N0     2048
#define N1     2048
#define N2     512

// ---------------------------------------------------------------------------
// Pack W (row-major [N][K], fp32) into fp16 WMMA B-operand tiles.
// Tile block = (ntile, kchunk): 32 lanes x 16 halves contiguous (1KB).
// Lane L (L<16): column n = 16*ntile + L, halves 0..7  = k 0..7,  8..15 = k 16..23
// Lane L (L>=16): same column group,      halves 0..7  = k 8..15, 8..15 = k 24..31
// (mirrors the documented 16-bit A-matrix lane layout).
__global__ void pack_w(const float* __restrict__ W, _Float16* __restrict__ P,
                       int N, int K) {
  int tid = blockIdx.x * blockDim.x + threadIdx.x;
  int kcN = K >> 5;
  int nblk = (N >> 4) * kcN;
  if (tid >= nblk * 32) return;
  int lane = tid & 31;
  int blk  = tid >> 5;
  int kc = blk % kcN;
  int nt = blk / kcN;
  int n  = nt * 16 + (lane & 15);
  int hi = lane >> 4;
  const float* src = W + (size_t)n * K + kc * 32 + hi * 8;
  _Float16* dst = P + (size_t)blk * 512 + lane * 16;
#pragma unroll
  for (int h = 0; h < 8; ++h) dst[h]     = (_Float16)src[h];
#pragma unroll
  for (int h = 0; h < 8; ++h) dst[8 + h] = (_Float16)src[16 + h];
}

// ---------------------------------------------------------------------------
// Poisson encoding for all T steps: spk[t,b,f] = (noise < clip(x,0,1)) ? 1 : 0
__global__ void encode(const float* __restrict__ x, const float* __restrict__ noise,
                       _Float16* __restrict__ spk, int total) {
  int i = blockIdx.x * blockDim.x + threadIdx.x;
  if (i >= total) return;
  int f = i % NIN;
  int b = (i / NIN) % BATCH;
  float r = x[b * NIN + f];
  r = fminf(fmaxf(r, 0.0f), 1.0f);
  spk[i] = (_Float16)((noise[i] < r) ? 1.0f : 0.0f);
}

__global__ void zero_f(float* __restrict__ p, int n) {
  int i = blockIdx.x * blockDim.x + threadIdx.x;
  if (i < n) p[i] = 0.0f;
}

// ---------------------------------------------------------------------------
// One wave computes one 16x16 output tile of I = spikes @ W^T + b, then runs
// the LIF update in-register on the WMMA accumulator (C layout: VGPR r holds
// M = r (lanes 0-15) / r+8 (lanes 16-31), N = lane&15).
template <int K, int N>
__global__ void __launch_bounds__(256)
lif_gemm(const _Float16* __restrict__ A,    // spikes in  [32][K]
         const _Float16* __restrict__ Bp,   // packed W   [N/16][K/32][512]
         const float* __restrict__ bias,    // [N]
         float* __restrict__ syn, float* __restrict__ mem, // [32][N]
         _Float16* __restrict__ spk_out,    // [32][N] or nullptr (last layer)
         float* __restrict__ acc,           // [32][N] (last layer) or nullptr
         float* __restrict__ count) {
  const int lane  = threadIdx.x & 31;
  const int wave  = threadIdx.x >> 5;
  const int gtile = blockIdx.x * 8 + wave;   // grid sized so all tiles valid
  const int mtile = gtile & 1;               // batch 0-15 / 16-31
  const int ntile = gtile >> 1;
  const int hi    = lane >> 4;
  const int l15   = lane & 15;

  const _Float16* arow = A + (size_t)(mtile * 16 + l15) * K + hi * 8;
  const _Float16* bptr = Bp + (size_t)ntile * (K >> 5) * 512 + lane * 16;

  v8f c = {};
  constexpr int kcN = K >> 5;                // 64: compile-time trip count
#pragma unroll 8
  for (int kc = 0; kc < kcN; ++kc) {
    union { v16h v; v8h h[2]; } a;
    a.h[0] = *(const v8h*)(arow);            // k  0..7  (or 8..15 for hi lanes)
    a.h[1] = *(const v8h*)(arow + 16);       // k 16..23 (or 24..31)
    v16h b = *(const v16h*)(bptr);           // contiguous packed B tile slice
    c = __builtin_amdgcn_wmma_f32_16x16x32_f16(
        /*neg_a=*/false, a.v, /*neg_b=*/false, b,
        /*c_mod=*/(short)0, c, /*reuse_a=*/false, /*reuse_b=*/false);
    arow += 32;
    bptr += 512;
  }

  // Fused LIF epilogue: syn' = 0.8*syn + I ; mem' = 0.95*mem + syn' ;
  // spike = mem' >= 1 ; reset fired ; clamp negatives.
  const int n = ntile * 16 + l15;
  const float bn = bias[n];
  const int batch0 = mtile * 16 + hi * 8;
  float cnt = 0.0f;
#pragma unroll
  for (int r = 0; r < 8; ++r) {
    int idx = (batch0 + r) * N + n;
    float I = c[r] + bn;
    float s = syn[idx] * 0.8f + I;           // TAU_SYN = 5
    float m = mem[idx] * 0.95f + s;          // TAU_MEM = 20
    float spike = (m >= 1.0f) ? 1.0f : 0.0f;
    m *= (1.0f - spike);
    m = (m > 0.0f) ? m : 0.0f;
    syn[idx] = s;
    mem[idx] = m;
    if (spk_out) spk_out[idx] = (_Float16)spike;
    if (acc)     acc[idx] += spike;          // unique tile ownership: no atomics
    cnt += spike;
  }
  // wave32 reduction of integer-valued spike count (exact in f32 < 2^24)
  for (int off = 16; off > 0; off >>= 1) cnt += __shfl_xor(cnt, off, 32);
  if (lane == 0) atomicAdd(count, cnt);
}

__global__ void finalize(const float* __restrict__ acc,
                         const float* __restrict__ count,
                         float* __restrict__ out) {
  int i = blockIdx.x * blockDim.x + threadIdx.x;
  if (i < BATCH * N2) out[i] = acc[i] * (1.0f / (float)TSTEPS);
  if (i == 0) out[BATCH * N2] = *count;
}

// ---------------------------------------------------------------------------
extern "C" void kernel_launch(void* const* d_in, const int* in_sizes, int n_in,
                              void* d_out, int out_size, void* d_ws, size_t ws_size,
                              hipStream_t stream) {
  (void)in_sizes; (void)n_in; (void)out_size; (void)ws_size;
  const float* x     = (const float*)d_in[0];
  const float* noise = (const float*)d_in[1];
  // d_in[2] = time_steps (int scalar) -> compile-time TSTEPS
  const float* W0 = (const float*)d_in[3];
  const float* b0 = (const float*)d_in[4];
  const float* W1 = (const float*)d_in[5];
  const float* b1 = (const float*)d_in[6];
  const float* W2 = (const float*)d_in[7];
  const float* b2 = (const float*)d_in[8];
  float* out = (float*)d_out;

  char* ws = (char*)d_ws;
  size_t off = 0;
  auto wsalloc = [&](size_t bytes) -> void* {
    void* p = ws + off;
    off += (bytes + 255) & ~(size_t)255;
    return p;
  };
  _Float16* wp0    = (_Float16*)wsalloc((size_t)N0 * NIN * 2);
  _Float16* wp1    = (_Float16*)wsalloc((size_t)N1 * N0 * 2);
  _Float16* wp2    = (_Float16*)wsalloc((size_t)N2 * N1 * 2);
  _Float16* spk_in = (_Float16*)wsalloc((size_t)TSTEPS * BATCH * NIN * 2);
  _Float16* spk1   = (_Float16*)wsalloc((size_t)BATCH * N0 * 2);
  _Float16* spk2   = (_Float16*)wsalloc((size_t)BATCH * N1 * 2);
  const int stateN = 2 * (BATCH * N0 + BATCH * N1 + BATCH * N2) + BATCH * N2 + 1;
  float* state = (float*)wsalloc((size_t)stateN * 4);
  float* syn0 = state;
  float* mem0 = syn0 + BATCH * N0;
  float* syn1 = mem0 + BATCH * N0;
  float* mem1 = syn1 + BATCH * N1;
  float* syn2 = mem1 + BATCH * N1;
  float* mem2 = syn2 + BATCH * N2;
  float* acc  = mem2 + BATCH * N2;
  float* cnt  = acc  + BATCH * N2;

  // Per-call init (harness does not re-poison between replays).
  zero_f<<<(stateN + 255) / 256, 256, 0, stream>>>(state, stateN);

  // One-time (per call) weight packing + spike encoding.
  pack_w<<<((N0 / 16) * (NIN / 32) * 32 + 255) / 256, 256, 0, stream>>>(W0, wp0, N0, NIN);
  pack_w<<<((N1 / 16) * (N0  / 32) * 32 + 255) / 256, 256, 0, stream>>>(W1, wp1, N1, N0);
  pack_w<<<((N2 / 16) * (N1  / 32) * 32 + 255) / 256, 256, 0, stream>>>(W2, wp2, N2, N1);
  {
    int total = TSTEPS * BATCH * NIN;
    encode<<<(total + 255) / 256, 256, 0, stream>>>(x, noise, spk_in, total);
  }

  // Sequential LIF chain: stream ordering provides cross-layer/step sync.
  for (int t = 0; t < TSTEPS; ++t) {
    lif_gemm<NIN, N0><<<N0 / 64, 256, 0, stream>>>(
        spk_in + (size_t)t * BATCH * NIN, wp0, b0, syn0, mem0, spk1, nullptr, cnt);
    lif_gemm<N0, N1><<<N1 / 64, 256, 0, stream>>>(
        spk1, wp1, b1, syn1, mem1, spk2, nullptr, cnt);
    lif_gemm<N1, N2><<<N2 / 64, 256, 0, stream>>>(
        spk2, wp2, b2, syn2, mem2, nullptr, acc, cnt);
  }

  finalize<<<(BATCH * N2 + 255) / 256, 256, 0, stream>>>(acc, cnt, out);
}
